// CVNetRerank_69604239999326
// MI455X (gfx1250) — matrix operand order repaired and checked by reading, compiled, not run on hardware
//
#include <hip/hip_runtime.h>
#include <hip/hip_bf16.h>

// ---------------- CDNA5 WMMA types ----------------
typedef __attribute__((ext_vector_type(16))) _Float16 v16h;
typedef __attribute__((ext_vector_type(8)))  _Float16 v8h;
typedef __attribute__((ext_vector_type(8)))  float    v8f;

#define B_  16
#define C_  1024
#define H_  16
#define W_  16
#define RC_ 256

__device__ __forceinline__ void lin_coef(int i, int n_out, int n_in,
                                         int& i0, int& i1, float& w) {
  float pos = (n_out > 1) ? (float)i * (float)(n_in - 1) / (float)(n_out - 1) : 0.0f;
  int f = (int)floorf(pos);
  if (f < 0) f = 0;
  if (f > n_in - 1) f = n_in - 1;
  i0 = f;
  i1 = (f + 1 < n_in) ? f + 1 : n_in - 1;
  w  = pos - (float)f;
}

// ---- zero-fill f16 scratch (zero page for branch-free conv padding) ----
__global__ void zero_f16_kernel(_Float16* __restrict__ p, int n) {
  int i = blockIdx.x * blockDim.x + threadIdx.x;
  if (i < n) p[i] = (_Float16)0.0f;
}

// ---- bilinear resize (B,C,16,16) f32 -> (B,sh,sw,C) f16 channels-last ----
__global__ void resize_f16_cl(const float* __restrict__ in,
                              _Float16* __restrict__ out,
                              int sh, int sw) {
  int idx = blockIdx.x * blockDim.x + threadIdx.x;
  int total = B_ * sh * sw * C_;
  if (idx >= total) return;
  int c = idx & (C_ - 1);
  int r = idx >> 10;
  int x = r % sw; r /= sw;
  int y = r % sh;
  int b = r / sh;
  int y0, y1, x0, x1; float wy, wx;
  lin_coef(y, sh, H_, y0, y1, wy);
  lin_coef(x, sw, W_, x0, x1, wx);
  const float* pc = in + ((size_t)(b * C_ + c)) * (H_ * W_);
  float v = (1.f - wy) * ((1.f - wx) * pc[y0 * W_ + x0] + wx * pc[y0 * W_ + x1])
          +        wy  * ((1.f - wx) * pc[y1 * W_ + x0] + wx * pc[y1 * W_ + x1]);
  out[idx] = (_Float16)v;
}

// ---- repack conv weight (256,1024,3,3) -> f16 A2[m][k], k = tap*1024 + c ----
__global__ void repack_w_f16(const float* __restrict__ w,
                             _Float16* __restrict__ wT) {
  int idx = blockIdx.x * blockDim.x + threadIdx.x;
  if (idx >= RC_ * 9216) return;
  int k = idx % 9216;
  int m = idx / 9216;
  int tap = k >> 10;          // 0..8 (kh*3+kw)
  int c   = k & 1023;
  wT[idx] = (_Float16)w[(size_t)m * 9216 + c * 9 + tap];
}

// ---- implicit-GEMM 3x3 conv, WMMA f16->f32, channels-last activations ----
// Wave computes a 16(M) x 64(N) tile: A fragment reused across 4 WMMAs.
// Padding handled branch-free via zero-page base pointers (hoisted per tap).
// out[(b*P + p)*256 + m], P = sh*sw, K = 9*1024 = 9216
__global__ __launch_bounds__(32) void conv3x3_wmma(
    const _Float16* __restrict__ in,   // (B, sh, sw, 1024) f16
    const _Float16* __restrict__ wT,   // (256, 9216) f16, A2[m][k]
    const _Float16* __restrict__ zpg,  // 2048 zero halfs
    float* __restrict__ out,           // (B*P, 256) f32
    int sh, int sw) {
  const int lane  = threadIdx.x;
  const int hlf   = lane >> 4;        // 0 / 1
  const int lm    = lane & 15;
  const int mTile = blockIdx.x;       // 16 tiles of M=256
  const int nTile = blockIdx.y;       // tiles of 64 positions
  const int P = sh * sw;

  // decode the 4 sub-tile positions this lane contributes B columns for
  int yy[4], xx[4], bb[4], nn[4];
  #pragma unroll
  for (int j = 0; j < 4; j++) {
    int n = nTile * 64 + j * 16 + lm;
    nn[j] = n;
    int b = n / P;
    int p = n - b * P;
    bb[j] = b;
    yy[j] = p / sw;
    xx[j] = p - yy[j] * sw;
  }

  const int m = mTile * 16 + lm;      // A-frag row
  const _Float16* arow = wT + (size_t)m * 9216;

  v8f acc0 = {}, acc1 = {}, acc2 = {}, acc3 = {};
  for (int tap = 0; tap < 9; tap++) {
    const int dy = tap / 3 - 1;
    const int dx = tap % 3 - 1;
    // per-tap base pointers: real pixel row or zero page (branch-free inner loop)
    const _Float16* bsrc0;
    const _Float16* bsrc1;
    const _Float16* bsrc2;
    const _Float16* bsrc3;
    {
      int iy, ix;
      iy = yy[0] + dy; ix = xx[0] + dx;
      bsrc0 = ((unsigned)iy < (unsigned)sh && (unsigned)ix < (unsigned)sw)
            ? in + (((size_t)(bb[0] * sh + iy) * sw + ix) << 10) : zpg;
      iy = yy[1] + dy; ix = xx[1] + dx;
      bsrc1 = ((unsigned)iy < (unsigned)sh && (unsigned)ix < (unsigned)sw)
            ? in + (((size_t)(bb[1] * sh + iy) * sw + ix) << 10) : zpg;
      iy = yy[2] + dy; ix = xx[2] + dx;
      bsrc2 = ((unsigned)iy < (unsigned)sh && (unsigned)ix < (unsigned)sw)
            ? in + (((size_t)(bb[2] * sh + iy) * sw + ix) << 10) : zpg;
      iy = yy[3] + dy; ix = xx[3] + dx;
      bsrc3 = ((unsigned)iy < (unsigned)sh && (unsigned)ix < (unsigned)sw)
            ? in + (((size_t)(bb[3] * sh + iy) * sw + ix) << 10) : zpg;
    }
    const _Float16* ap = arow + tap * 1024;
    const int co = hlf * 16;
    for (int c = 0; c < 1024; c += 32) {
      v16h af;
      af = *(const v16h*)(ap + c + hlf * 16);
      __builtin_prefetch(ap + c + 256, 0, 1);     // global_prefetch_b8
      v16h b0 = *(const v16h*)(bsrc0 + c + co);
      v16h b1 = *(const v16h*)(bsrc1 + c + co);
      v16h b2 = *(const v16h*)(bsrc2 + c + co);
      v16h b3 = *(const v16h*)(bsrc3 + c + co);
      acc0 = __builtin_amdgcn_wmma_f32_16x16x32_f16(false, af, false, b0, (short)0, acc0, false, false);
      acc1 = __builtin_amdgcn_wmma_f32_16x16x32_f16(false, af, false, b1, (short)0, acc1, false, false);
      acc2 = __builtin_amdgcn_wmma_f32_16x16x32_f16(false, af, false, b2, (short)0, acc2, false, false);
      acc3 = __builtin_amdgcn_wmma_f32_16x16x32_f16(false, af, false, b3, (short)0, acc3, false, false);
    }
  }
  // D layout: lanes 0-15 -> M=i, lanes 16-31 -> M=8+i, col = lm
  const int mo = mTile * 16 + hlf * 8;
  *(v8f*)(out + (size_t)nn[0] * 256 + mo) = acc0;
  *(v8f*)(out + (size_t)nn[1] * 256 + mo) = acc1;
  *(v8f*)(out + (size_t)nn[2] * 256 + mo) = acc2;
  *(v8f*)(out + (size_t)nn[3] * 256 + mo) = acc3;
}

// ---- per-position L2 norm over 256 channels, f32 -> normalized f16 ----
__global__ __launch_bounds__(32) void l2norm_f16(const float* __restrict__ conv,
                                                 _Float16* __restrict__ outn) {
  const int p = blockIdx.x;
  const int lane = threadIdx.x;
  const float* row = conv + (size_t)p * 256 + lane * 8;
  v8f v = *(const v8f*)row;
  float s = 0.f;
  #pragma unroll
  for (int i = 0; i < 8; i++) s += v[i] * v[i];
  #pragma unroll
  for (int off = 16; off > 0; off >>= 1) s += __shfl_xor(s, off);
  float inv = 1.0f / (sqrtf(s) + 1e-5f);
  _Float16* o = outn + (size_t)p * 256 + lane * 8;
  #pragma unroll
  for (int i = 0; i < 8; i++) o[i] = (_Float16)(v[i] * inv);
}

// ---- correlation GEMM: corr[b][pq][pk] = sum_c qn[b][pq][c]*kn[b][pk][c] ----
__global__ __launch_bounds__(32) void corr_wmma(
    const _Float16* __restrict__ qn,   // (B, Pq, 256)
    const _Float16* __restrict__ kn,   // (B, Pk, 256)
    float* __restrict__ corr,          // (B, Pq, Pk)
    int Pq, int Pk) {
  const int lane = threadIdx.x;
  const int hlf  = lane >> 4;
  const int lm   = lane & 15;
  const int mTile = blockIdx.x;
  const int nTile = blockIdx.y;
  const int b     = blockIdx.z;
  const _Float16* arow = qn + ((size_t)b * Pq + mTile * 16 + lm) * 256;
  const _Float16* brow = kn + ((size_t)b * Pk + nTile * 16 + lm) * 256;
  v8f acc = {};
  #pragma unroll
  for (int kb = 0; kb < 256; kb += 32) {
    v16h af, bf;
    af = *(const v16h*)(arow + kb + hlf * 16);
    bf = *(const v16h*)(brow + kb + hlf * 16);
    acc = __builtin_amdgcn_wmma_f32_16x16x32_f16(false, af, false, bf,
                                                 (short)0, acc, false, false);
  }
  const int n = nTile * 16 + lm;
  float* base = corr + ((size_t)b * Pq + mTile * 16 + hlf * 8) * Pk + n;
  #pragma unroll
  for (int i = 0; i < 8; i++) base[(size_t)i * Pk] = acc[i];
}

// ---- corr bilinear resize stage 1: (B,sq,sq,sk,sk) -> (B,16,16,sk,sk) ----
__global__ void corr_resize_q(const float* __restrict__ corr,
                              float* __restrict__ tmp, int sq, int sk) {
  int U = sk * sk;
  int idx = blockIdx.x * blockDim.x + threadIdx.x;
  int total = B_ * 256 * U;
  if (idx >= total) return;
  int u = idx % U;
  int r = idx / U;
  int X = r & 15; r >>= 4;
  int Y = r & 15;
  int b = r >> 4;
  int y0, y1, x0, x1; float wy, wx;
  lin_coef(Y, 16, sq, y0, y1, wy);
  lin_coef(X, 16, sq, x0, x1, wx);
  const float* pb = corr + (size_t)b * sq * sq * U;
  float v = (1.f - wy) * ((1.f - wx) * pb[(y0 * sq + x0) * U + u] + wx * pb[(y0 * sq + x1) * U + u])
          +        wy  * ((1.f - wx) * pb[(y1 * sq + x0) * U + u] + wx * pb[(y1 * sq + x1) * U + u]);
  tmp[idx] = v;
}

// ---- stage 2 + relu + write into x[b][pair][Y][X][U][V] ----
__global__ void corr_resize_k_relu(const float* __restrict__ tmp,
                                   float* __restrict__ x, int sk, int pair) {
  int idx = blockIdx.x * blockDim.x + threadIdx.x;
  if (idx >= B_ * 65536) return;
  int V = idx & 15; int r = idx >> 4;
  int U = r & 15;   r >>= 4;
  int X = r & 15;   r >>= 4;
  int Y = r & 15;
  int b = r >> 4;
  int u0, u1, v0, v1; float wu, wv;
  lin_coef(U, 16, sk, u0, u1, wu);
  lin_coef(V, 16, sk, v0, v1, wv);
  const float* pb = tmp + ((size_t)b * 256 + Y * 16 + X) * sk * sk;
  float val = (1.f - wu) * ((1.f - wv) * pb[u0 * sk + v0] + wv * pb[u0 * sk + v1])
            +        wu  * ((1.f - wv) * pb[u1 * sk + v0] + wv * pb[u1 * sk + v1]);
  val = fmaxf(val, 0.f);
  x[((size_t)b * 9 + pair) * 65536 + (Y * 16 + X) * 256 + U * 16 + V] = val;
}

// ---- cp_conv4d: conv over a-dims, pivot = subsampled b-dims (writes) ----
__global__ void cp4d_convA(const float* __restrict__ xin, const float* __restrict__ w,
                           const float* __restrict__ bias, float* __restrict__ yout,
                           int ci, int co, int di, int dout, int ksz, int stride) {
  extern __shared__ float tile[];      // ci * di * di
  const int d2  = di * di;
  const int d2o = dout * dout;
  int blk = blockIdx.x;
  int wb = blk % dout; int t = blk / dout;
  int hb = t % dout;   int b = t / dout;
  int hbs = hb * stride, wbs = wb * stride;
  for (int i = threadIdx.x; i < ci * d2; i += blockDim.x)
    tile[i] = xin[((size_t)(b * ci) * d2 + i) * d2 + hbs * di + wbs];
  __syncthreads();
  int pad = ksz >> 1;
  for (int oi = threadIdx.x; oi < co * d2o; oi += blockDim.x) {
    int ox = oi % dout; int tt = oi / dout;
    int oy = tt % dout; int m = tt / dout;
    float acc = bias[m];
    const float* wm = w + (size_t)m * ci * ksz * ksz;
    for (int c = 0; c < ci; c++)
      for (int ky = 0; ky < ksz; ky++) {
        int iy = oy * stride + ky - pad;
        if ((unsigned)iy >= (unsigned)di) continue;
        for (int kx = 0; kx < ksz; kx++) {
          int ix = ox * stride + kx - pad;
          if ((unsigned)ix >= (unsigned)di) continue;
          acc += tile[c * d2 + iy * di + ix] * wm[(c * ksz + ky) * ksz + kx];
        }
      }
    yout[((size_t)((b * co + m) * dout + oy) * dout + ox) * d2o + hb * dout + wb] = acc;
  }
}

// ---- cp_conv4d: conv over b-dims, pivot = subsampled a-dims (accumulates) ----
__global__ void cp4d_convB(const float* __restrict__ xin, const float* __restrict__ w,
                           const float* __restrict__ bias, float* __restrict__ yout,
                           int ci, int co, int di, int dout, int ksz, int stride) {
  extern __shared__ float tile[];
  const int d2  = di * di;
  const int d2o = dout * dout;
  int blk = blockIdx.x;
  int wa = blk % dout; int t = blk / dout;
  int ha = t % dout;   int b = t / dout;
  int has = ha * stride, was = wa * stride;
  for (int i = threadIdx.x; i < ci * d2; i += blockDim.x) {
    int c = i / d2; int j = i - c * d2;
    tile[i] = xin[((size_t)(b * ci + c) * d2 + has * di + was) * d2 + j];
  }
  __syncthreads();
  int pad = ksz >> 1;
  for (int oi = threadIdx.x; oi < co * d2o; oi += blockDim.x) {
    int ov = oi % dout; int tt = oi / dout;
    int ou = tt % dout; int m = tt / dout;
    float acc = bias[m];
    const float* wm = w + (size_t)m * ci * ksz * ksz;
    for (int c = 0; c < ci; c++)
      for (int ky = 0; ky < ksz; ky++) {
        int iu = ou * stride + ky - pad;
        if ((unsigned)iu >= (unsigned)di) continue;
        for (int kx = 0; kx < ksz; kx++) {
          int iv = ov * stride + kx - pad;
          if ((unsigned)iv >= (unsigned)di) continue;
          acc += tile[c * d2 + iu * di + iv] * wm[(c * ksz + ky) * ksz + kx];
        }
      }
    yout[((size_t)((b * co + m) * dout + ha) * dout + wa) * d2o + ou * dout + ov] += acc;
  }
}

// ---- group norm (groups=4) + relu, in place, x (B, co, spatial) ----
__global__ __launch_bounds__(256) void gn_relu(float* __restrict__ xio,
                                               const float* __restrict__ g,
                                               const float* __restrict__ bt,
                                               int co, int spatial) {
  __shared__ float rs[256], rq[256];
  int b  = blockIdx.x >> 2;
  int gr = blockIdx.x & 3;
  int cpg = co >> 2;
  size_t base = ((size_t)b * co + gr * cpg) * spatial;
  int n = cpg * spatial;
  float s = 0.f, q = 0.f;
  for (int i = threadIdx.x; i < n; i += blockDim.x) {
    float v = xio[base + i]; s += v; q += v * v;
  }
  rs[threadIdx.x] = s; rq[threadIdx.x] = q;
  __syncthreads();
  for (int off = 128; off > 0; off >>= 1) {
    if (threadIdx.x < off) {
      rs[threadIdx.x] += rs[threadIdx.x + off];
      rq[threadIdx.x] += rq[threadIdx.x + off];
    }
    __syncthreads();
  }
  float mu  = rs[0] / (float)n;
  float var = rq[0] / (float)n - mu * mu;
  if (var < 0.f) var = 0.f;
  float inv = rsqrtf(var + 1e-5f);
  for (int i = threadIdx.x; i < n; i += blockDim.x) {
    int c = gr * cpg + i / spatial;
    float v = (xio[base + i] - mu) * inv * g[c] + bt[c];
    xio[base + i] = fmaxf(v, 0.f);
  }
}

// ---- spatial mean: (B, co, spatial) -> feat (B, co) ----
__global__ void mean_sp(const float* __restrict__ xin, float* __restrict__ feat,
                        int co, int spatial) {
  int idx = blockIdx.x * blockDim.x + threadIdx.x;
  if (idx >= B_ * co) return;
  const float* p = xin + (size_t)idx * spatial;
  float s = 0.f;
  for (int i = 0; i < spatial; i++) s += p[i];
  feat[idx] = s / (float)spatial;
}

// ---- MLP head + softmax[:,1] ----
__global__ __launch_bounds__(64) void mlp_head(const float* __restrict__ feat,
                                               const float* __restrict__ w1,
                                               const float* __restrict__ b1,
                                               const float* __restrict__ w2,
                                               const float* __restrict__ b2,
                                               float* __restrict__ out) {
  __shared__ float h[64];
  int b = blockIdx.x, t = threadIdx.x;
  const float* f = feat + b * 64;
  float s = b1[t];
  for (int c = 0; c < 64; c++) s += f[c] * w1[t * 64 + c];
  h[t] = fmaxf(s, 0.f);
  __syncthreads();
  if (t == 0) {
    float l0 = b2[0], l1 = b2[1];
    for (int c = 0; c < 64; c++) { l0 += h[c] * w2[c]; l1 += h[c] * w2[64 + c]; }
    float mx = fmaxf(l0, l1);
    float e0 = expf(l0 - mx), e1 = expf(l1 - mx);
    out[b] = e1 / (e0 + e1);
  }
}

// =====================================================================
extern "C" void kernel_launch(void* const* d_in, const int* in_sizes, int n_in,
                              void* d_out, int out_size, void* d_ws, size_t ws_size,
                              hipStream_t stream) {
  (void)in_sizes; (void)n_in; (void)out_size; (void)ws_size;
  // Input layout (setup_inputs dict order, params flattened in insertion order):
  // 0: query_features, 1: key_features, 2-4: conv2d_w0..2,
  // 5+6*li: w1, w2, b1, b2, gn_g, gn_b per layer (li=0..5),
  // 41: mlp_w1, 42: mlp_b1, 43: mlp_w2, 44: mlp_b2
  const float* qf = (const float*)d_in[0];
  const float* kf = (const float*)d_in[1];
  const float* convw[3] = {(const float*)d_in[2], (const float*)d_in[3], (const float*)d_in[4]};

  const int shs[3] = {4, 8, 16};
  const int Ps[3]  = {16, 64, 256};

  // deterministic workspace carving
  char* wp = (char*)d_ws;
  auto alloc = [&](size_t bytes) -> void* {
    void* r = (void*)wp;
    wp += (bytes + 255) & ~(size_t)255;
    return r;
  };
  _Float16 *qr[3], *kr[3], *wT[3], *qn[3], *kn[3];
  for (int s = 0; s < 3; s++) qr[s] = (_Float16*)alloc((size_t)B_ * Ps[s] * C_ * 2);
  for (int s = 0; s < 3; s++) kr[s] = (_Float16*)alloc((size_t)B_ * Ps[s] * C_ * 2);
  for (int s = 0; s < 3; s++) wT[s] = (_Float16*)alloc((size_t)RC_ * 9216 * 2);
  for (int s = 0; s < 3; s++) qn[s] = (_Float16*)alloc((size_t)B_ * Ps[s] * RC_ * 2);
  for (int s = 0; s < 3; s++) kn[s] = (_Float16*)alloc((size_t)B_ * Ps[s] * RC_ * 2);
  _Float16* zpg = (_Float16*)alloc(2048 * 2);                    // zero page
  float* convf = (float*)alloc((size_t)B_ * 256 * RC_ * 4);      // reused per scale
  float* corr  = (float*)alloc((size_t)B_ * 256 * 256 * 4);
  float* ctmp  = (float*)alloc((size_t)B_ * 256 * 256 * 4);
  float* xbuf  = (float*)alloc((size_t)B_ * 9 * 65536 * 4);
  float* ya    = (float*)alloc((size_t)B_ * 16 * 4096 * 4);
  float* yb    = (float*)alloc((size_t)B_ * 16 * 4096 * 4);
  float* feat  = (float*)alloc((size_t)B_ * 64 * 4);

  // 0) zero page for conv padding (ws is poisoned; must re-zero every launch)
  zero_f16_kernel<<<8, 256, 0, stream>>>(zpg, 2048);

  // 1) resize q/k to 3 scales (f16 channels-last) + repack conv weights
  for (int s = 0; s < 3; s++) {
    int total = B_ * Ps[s] * C_;
    resize_f16_cl<<<total / 256, 256, 0, stream>>>(qf, qr[s], shs[s], shs[s]);
    resize_f16_cl<<<total / 256, 256, 0, stream>>>(kf, kr[s], shs[s], shs[s]);
    repack_w_f16<<<(RC_ * 9216) / 256, 256, 0, stream>>>(convw[s], wT[s]);
  }

  // 2) conv (WMMA implicit GEMM, 16x64 tile/wave) + l2norm -> normalized f16 feats
  for (int s = 0; s < 3; s++) {
    int NP = B_ * Ps[s];
    conv3x3_wmma<<<dim3(16, NP / 64), 32, 0, stream>>>(qr[s], wT[s], zpg, convf, shs[s], shs[s]);
    l2norm_f16<<<NP, 32, 0, stream>>>(convf, qn[s]);
    conv3x3_wmma<<<dim3(16, NP / 64), 32, 0, stream>>>(kr[s], wT[s], zpg, convf, shs[s], shs[s]);
    l2norm_f16<<<NP, 32, 0, stream>>>(convf, kn[s]);
  }

  // 3) 9 correlation GEMMs (WMMA) + separable bilinear resize + relu -> x
  for (int si = 0; si < 3; si++)
    for (int sj = 0; sj < 3; sj++) {
      int pair = si * 3 + sj;
      int Pq = Ps[si], Pk = Ps[sj], sq = shs[si], sk = shs[sj];
      corr_wmma<<<dim3(Pq / 16, Pk / 16, B_), 32, 0, stream>>>(qn[si], kn[sj], corr, Pq, Pk);
      int tot1 = B_ * 256 * sk * sk;
      corr_resize_q<<<(tot1 + 255) / 256, 256, 0, stream>>>(corr, ctmp, sq, sk);
      corr_resize_k_relu<<<(B_ * 65536) / 256, 256, 0, stream>>>(ctmp, xbuf, sk, pair);
    }

  // 4) six cp_conv4d layers + group-norm + relu
  const int Lci[6] = {9, 16, 16, 32, 32, 32};
  const int Lco[6] = {16, 16, 32, 32, 32, 64};
  const int Lk[6]  = {5, 3, 3, 3, 3, 3};
  const int Ls[6]  = {2, 1, 2, 1, 1, 2};
  const float* cur = xbuf;
  int di = 16;
  for (int li = 0; li < 6; li++) {
    int ci = Lci[li], co = Lco[li], ks = Lk[li], st = Ls[li];
    int dout = (di + st - 1) / st;
    const float* w1 = (const float*)d_in[5 + li * 6 + 0];
    const float* w2 = (const float*)d_in[5 + li * 6 + 1];
    const float* b1 = (const float*)d_in[5 + li * 6 + 2];
    const float* b2 = (const float*)d_in[5 + li * 6 + 3];
    const float* gg = (const float*)d_in[5 + li * 6 + 4];
    const float* gb = (const float*)d_in[5 + li * 6 + 5];
    float* nxt = (li & 1) ? yb : ya;
    size_t lds = (size_t)ci * di * di * sizeof(float);
    cp4d_convA<<<B_ * dout * dout, 256, lds, stream>>>(cur, w1, b1, nxt, ci, co, di, dout, ks, st);
    cp4d_convB<<<B_ * dout * dout, 256, lds, stream>>>(cur, w2, b2, nxt, ci, co, di, dout, ks, st);
    int spatial = dout * dout * dout * dout;
    gn_relu<<<B_ * 4, 256, 0, stream>>>(nxt, gg, gb, co, spatial);
    cur = nxt;
    di = dout;
  }

  // 5) mean over 4D spatial -> feat (B,64); MLP + softmax -> out[b]
  int spatial = di * di * di * di;     // 16
  mean_sp<<<(B_ * 64 + 255) / 256, 256, 0, stream>>>(cur, feat, 64, spatial);
  mlp_head<<<B_, 64, 0, stream>>>(feat,
                                  (const float*)d_in[41], (const float*)d_in[42],
                                  (const float*)d_in[43], (const float*)d_in[44],
                                  (float*)d_out);
}
